// MLAttention2_25744033972585
// MI455X (gfx1250) — matrix operand order repaired and checked by reading, not compiled, not run on hardware
//
#include <hip/hip_runtime.h>

// MLAttention: scores = W[L,H] x X[B,S,H]^T -> softmax(S) -> out = P x X
// B=8, S=512, H=256, L=32768.  Compute-bound (137 GFLOP vs ~292MB HBM), so we
// run both GEMMs on v_wmma_f32_16x16x32_bf16 (f32 accumulate), f32 softmax.

#define NUM_LABELS 32768
#define HIDDEN     256
#define BATCH      8
#define SEQ        512
#define LT         64      // labels per workgroup

// padded LDS row strides (bank-conflict avoidance, 64 banks x 4B)
#define SC_STRIDE  532     // floats  (512 + 20)
#define WT_STRIDE  264     // bf16    (256 + 8)
#define P_STRIDE   520     // bf16    (512 + 8)

typedef __attribute__((ext_vector_type(16))) __bf16       v16bf;
typedef __attribute__((ext_vector_type(8)))  float        v8f;
typedef __attribute__((ext_vector_type(4)))  unsigned int u32x4;
typedef __attribute__((ext_vector_type(8)))  unsigned int u32x8;

__device__ __forceinline__ unsigned short f32_to_bf16_rne(float f) {
  unsigned int u = __float_as_uint(f);
  unsigned int r = u + 0x7FFFu + ((u >> 16) & 1u);
  return (unsigned short)(r >> 16);
}

// A-matrix 16x32 bf16 fragment (ISA 7.12.2): lane half selects K base:
// lanes 0-15: K = kb+0..7 (v0-3) and kb+16..23 (v4-7); lanes 16-31: +8.
// Caller passes kb already including (lane>>4)*8.
__device__ __forceinline__ v16bf load_a_frag(const unsigned short* row, int kb) {
  u32x4 lo = *(const u32x4*)(row + kb);
  u32x4 hi = *(const u32x4*)(row + kb + 16);
  u32x8 t;
  t[0] = lo[0]; t[1] = lo[1]; t[2] = lo[2]; t[3] = lo[3];
  t[4] = hi[0]; t[5] = hi[1]; t[6] = hi[2]; t[7] = hi[3];
  return __builtin_bit_cast(v16bf, t);
}

// Contiguous-k B fragment (not ISA order; unused): lane = column N, lanes 0-15 hold K=kb..kb+15
// contiguous, lanes 16-31 hold K=kb+16..kb+31.  Caller passes kb already
// including (lane>>4)*16 -> a single contiguous 32B load.
__device__ __forceinline__ v16bf load_b_frag(const unsigned short* row, int kb) {
  return __builtin_bit_cast(v16bf, *(const u32x8*)(row + kb));
}

__global__ void prep_w_kernel(const float* __restrict__ w,
                              unsigned short* __restrict__ wbf, int n) {
  int i = blockIdx.x * blockDim.x + threadIdx.x;
  if (i < n) wbf[i] = f32_to_bf16_rne(w[i]);
}

__global__ void prep_x_kernel(const float* __restrict__ x,
                              unsigned short* __restrict__ xbf,
                              unsigned short* __restrict__ xt, int n) {
  int i = blockIdx.x * blockDim.x + threadIdx.x;
  if (i >= n) return;
  unsigned short v = f32_to_bf16_rne(x[i]);
  xbf[i] = v;                               // [B][S][H]
  int h = i & (HIDDEN - 1);
  int s = (i >> 8) & (SEQ - 1);             // H = 2^8
  int b = i >> 17;                          // S*H = 2^17
  xt[((b * HIDDEN) + h) * SEQ + s] = v;     // [B][H][S]
}

__global__ void __launch_bounds__(256)
mlattn_main_kernel(const unsigned short* __restrict__ wbf,   // [L][H] bf16
                   const unsigned short* __restrict__ xbf,   // [B][S][H] bf16
                   const unsigned short* __restrict__ xt,    // [B][H][S] bf16
                   float* __restrict__ out) {                // [B][L][H] f32
  extern __shared__ char smem[];
  float*          s_sc = (float*)smem;                                   // [LT][SC_STRIDE]
  unsigned short* s_wt = (unsigned short*)(smem + LT * SC_STRIDE * 4);   // [LT][WT_STRIDE]
  unsigned short* s_p  = (unsigned short*)(smem + LT * SC_STRIDE * 4
                                                + LT * WT_STRIDE * 2);   // [LT][P_STRIDE]

  const int b     = blockIdx.y;
  const int lbase = blockIdx.x * LT;
  const int tid   = threadIdx.x;
  const int wave  = tid >> 5;
  const int lane  = tid & 31;
  const int lhalf = lane >> 4;
  const int l15   = lane & 15;

  // ---- stage W tile (64x256 bf16) into LDS: 2048 16B chunks over 256 threads
  #pragma unroll
  for (int i = 0; i < 8; ++i) {
    int idx = tid + i * 256;
    int row = idx >> 5;              // 32 chunks of 8 bf16 per row
    int c8  = (idx & 31) << 3;
    *(u32x4*)(s_wt + row * WT_STRIDE + c8) =
        *(const u32x4*)(wbf + (long)(lbase + row) * HIDDEN + c8);
  }
  __syncthreads();

  // ---- Pass 1: scores[64 x 64-chunk] per wave; 4 M-tiles x 4 S-tiles
  {
    const int sw = wave * 64;
    const unsigned short* xb = xbf + (long)b * SEQ * HIDDEN;
    v8f acc[4][4] = {};
    #pragma unroll
    for (int it = 0; it < HIDDEN / 32; ++it) {
      const int kbA = it * 32 + lhalf * 8;
      const int kbB = it * 32 + lhalf * 8;
      v16bf a[4], bb[4];
      #pragma unroll
      for (int m = 0; m < 4; ++m)
        a[m] = load_a_frag(s_wt + (m * 16 + l15) * WT_STRIDE, kbA);
      #pragma unroll
      for (int st = 0; st < 4; ++st)
        bb[st] = load_a_frag(xb + (long)(sw + st * 16 + l15) * HIDDEN, kbB);
      #pragma unroll
      for (int m = 0; m < 4; ++m)
        #pragma unroll
        for (int st = 0; st < 4; ++st)
          acc[m][st] = __builtin_amdgcn_wmma_f32_16x16x32_bf16(
              false, a[m], false, bb[st], (short)0, acc[m][st], false, false);
    }
    // D layout: lane=N column, VGPR v = row v (+8 for upper lane half)
    #pragma unroll
    for (int m = 0; m < 4; ++m)
      #pragma unroll
      for (int st = 0; st < 4; ++st)
        #pragma unroll
        for (int v = 0; v < 8; ++v) {
          int row = m * 16 + v + lhalf * 8;
          int col = sw + st * 16 + l15;
          s_sc[row * SC_STRIDE + col] = acc[m][st][v];
        }
  }
  __syncthreads();

  // ---- softmax over S=512 per label row; 4 threads per row
  {
    const int row = tid >> 2;
    const int j   = tid & 3;
    float* sr = s_sc + row * SC_STRIDE;
    float mx = -3.402823466e38f;
    #pragma unroll 8
    for (int i = 0; i < 128; ++i) mx = fmaxf(mx, sr[i * 4 + j]);
    mx = fmaxf(mx, __shfl_xor(mx, 1, 4));
    mx = fmaxf(mx, __shfl_xor(mx, 2, 4));
    float sum = 0.f;
    #pragma unroll 8
    for (int i = 0; i < 128; ++i) {
      float e = __expf(sr[i * 4 + j] - mx);
      sr[i * 4 + j] = e;                 // each thread touches only its own cols
      sum += e;
    }
    sum += __shfl_xor(sum, 1, 4);
    sum += __shfl_xor(sum, 2, 4);
    float inv = 1.0f / sum;
    unsigned short* pr = s_p + row * P_STRIDE;
    #pragma unroll 8
    for (int i = 0; i < 128; ++i)
      pr[i * 4 + j] = f32_to_bf16_rne(sr[i * 4 + j] * inv);
  }
  __syncthreads();

  // ---- Pass 2: out[64 x 32-chunk] per wave; 4 M-tiles x 2 H-tiles, K = S
  {
    const int hw = wave * 32;
    const unsigned short* xtb = xt + (long)b * HIDDEN * SEQ;
    v8f acc[4][2] = {};
    #pragma unroll
    for (int it = 0; it < SEQ / 32; ++it) {
      const int kbA = it * 32 + lhalf * 8;
      const int kbB = it * 32 + lhalf * 8;
      v16bf a[4], bb[2];
      #pragma unroll
      for (int m = 0; m < 4; ++m)
        a[m] = load_a_frag(s_p + (m * 16 + l15) * P_STRIDE, kbA);
      #pragma unroll
      for (int ht = 0; ht < 2; ++ht)
        bb[ht] = load_a_frag(xtb + (long)(hw + ht * 16 + l15) * SEQ, kbB);
      #pragma unroll
      for (int m = 0; m < 4; ++m)
        #pragma unroll
        for (int ht = 0; ht < 2; ++ht)
          acc[m][ht] = __builtin_amdgcn_wmma_f32_16x16x32_bf16(
              false, a[m], false, bb[ht], (short)0, acc[m][ht], false, false);
    }
    float* ob = out + ((long)b * NUM_LABELS + lbase) * HIDDEN;
    #pragma unroll
    for (int m = 0; m < 4; ++m)
      #pragma unroll
      for (int ht = 0; ht < 2; ++ht)
        #pragma unroll
        for (int v = 0; v < 8; ++v) {
          int row = m * 16 + v + lhalf * 8;
          int h   = hw + ht * 16 + l15;
          ob[(long)row * HIDDEN + h] = acc[m][ht][v];
        }
  }
}

extern "C" void kernel_launch(void* const* d_in, const int* in_sizes, int n_in,
                              void* d_out, int out_size, void* d_ws, size_t ws_size,
                              hipStream_t stream) {
  const float* x = (const float*)d_in[0];   // inputs [8,512,256] f32
  const float* w = (const float*)d_in[1];   // attn_weight [32768,256] f32
  float* out = (float*)d_out;               // [8,32768,256] f32

  unsigned short* wbf = (unsigned short*)d_ws;                       // 16 MB
  unsigned short* xbf = wbf + (size_t)NUM_LABELS * HIDDEN;           // 2 MB
  unsigned short* xt  = xbf + (size_t)BATCH * SEQ * HIDDEN;          // 2 MB

  const int nw = NUM_LABELS * HIDDEN;
  prep_w_kernel<<<(nw + 255) / 256, 256, 0, stream>>>(w, wbf, nw);
  const int nx = BATCH * SEQ * HIDDEN;
  prep_x_kernel<<<(nx + 255) / 256, 256, 0, stream>>>(x, xbf, xt, nx);

  const size_t smem = (size_t)LT * SC_STRIDE * 4 +
                      (size_t)LT * WT_STRIDE * 2 +
                      (size_t)LT * P_STRIDE * 2;   // ~231 KB of the 320 KB LDS
  dim3 grid(NUM_LABELS / LT, BATCH);
  mlattn_main_kernel<<<grid, 256, smem, stream>>>(wbf, xbf, xt, out);
}
